// CrossModalityAttention_5738076308098
// MI455X (gfx1250) — matrix operand order repaired and checked by reading, not compiled, not run on hardware
//
#include <hip/hip_runtime.h>

#define DIM    1024
#define HEADS  16
#define HDIM   64
#define BB     2
#define TT     2048
#define LTOT   4096   // T + S key rows
#define ROWS_X 4096   // B*T rows of x
#define ROWS_ALL 8192 // x rows + context rows

typedef __attribute__((ext_vector_type(16))) __bf16 v16bf;
typedef __attribute__((ext_vector_type(8)))  float  v8f;
typedef unsigned int u32x4 __attribute__((ext_vector_type(4)));
typedef int          i32x4 __attribute__((ext_vector_type(4)));
typedef int          i32x8 __attribute__((ext_vector_type(8)));

union ABu { v16bf v; u32x4 u[2]; };

static __device__ __forceinline__ v8f wmma_bf16(v16bf a, v16bf b, v8f c) {
  // (neg_a, A, neg_b, B, c_mod, C, reuse_a, reuse_b)
  return __builtin_amdgcn_wmma_f32_16x16x32_bf16(false, a, false, b, (short)0, c, false, false);
}

// Generic shared pointer -> LDS byte offset (low 32 bits of the address)
static __device__ __forceinline__ unsigned int lds_off_of(const void* p) {
  return (unsigned int)(unsigned long long)p;
}

// ---------------------------------------------------------------------------
// Tensor Data Mover: async 2D tile (2-byte elements) global -> LDS.
// D# per CDNA5 ISA ch.8: group0 = {flags, lds_addr, global_addr, type=2},
// group1 = {data_size=2B, tensor_dim0/1, tile_dim0/1, dim0 stride},
// groups 2/3 zero (2D tensor). Tracked by TENSORcnt.
// ---------------------------------------------------------------------------
static __device__ __forceinline__ void tdm_load_2d(
    unsigned int lds_addr, const void* gptr,
    unsigned int tile_w,   unsigned int tile_h,    // in elements / rows
    unsigned int tensor_w, unsigned int tensor_h,  // full extents (elements)
    unsigned int row_stride)                       // elements between rows
{
  unsigned long long ga = (unsigned long long)gptr;
  u32x4 g0;
  g0[0] = 1u;                                   // count=1 (valid user D#)
  g0[1] = lds_addr;                             // lds_addr [63:32]
  g0[2] = (unsigned int)(ga & 0xffffffffu);     // global_addr [95:64]
  g0[3] = (unsigned int)((ga >> 32) & 0x01ffffffu) | (2u << 30); // addr hi | type=2
  i32x8 g1;
  g1[0] = (int)(1u << 16);                      // data_size=1 (2 bytes/elem)
  g1[1] = (int)((tensor_w & 0xffffu) << 16);    // tensor_dim0 lo
  g1[2] = (int)(((tensor_w >> 16) & 0xffffu) | ((tensor_h & 0xffffu) << 16));
  g1[3] = (int)(((tensor_h >> 16) & 0xffffu) | (tile_w << 16));  // tile_dim0
  g1[4] = (int)(tile_h & 0xffffu);              // tile_dim1 (tile_dim2=0)
  g1[5] = (int)row_stride;                      // tensor_dim0_stride lo
  g1[6] = 0;                                    // stride hi, dim1_stride lo
  g1[7] = 0;
  i32x4 gz = {0, 0, 0, 0};
#if defined(__clang_major__) && __clang_major__ >= 23
  i32x8 gz8 = {0, 0, 0, 0, 0, 0, 0, 0};
  __builtin_amdgcn_tensor_load_to_lds(g0, g1, gz, gz, gz8, 0);
#else
  __builtin_amdgcn_tensor_load_to_lds(g0, g1, gz, gz, 0);
#endif
}

// ---------------------------------------------------------------------------
// Kernel 1: LayerNorm (x rows then context rows) -> bf16 normalized activations
// ---------------------------------------------------------------------------
__global__ __launch_bounds__(256)
void ln_bf16_kernel(const float* __restrict__ x, const float* __restrict__ ctx,
                    const float* __restrict__ g, const float* __restrict__ beta,
                    __bf16* __restrict__ xn) {
  int row = blockIdx.x; // 0..8191
  const float* src = (row < ROWS_X) ? (x + (size_t)row * DIM)
                                    : (ctx + (size_t)(row - ROWS_X) * DIM);
  __shared__ float red[2][8];
  int tid = threadIdx.x;
  float s = 0.f, ss = 0.f;
  float vals[4];
#pragma unroll
  for (int i = 0; i < 4; ++i) {
    float v = src[tid + 256 * i];
    vals[i] = v;
    s += v; ss += v * v;
  }
#pragma unroll
  for (int m = 1; m < 32; m <<= 1) {
    s  += __shfl_xor(s,  m, 32);
    ss += __shfl_xor(ss, m, 32);
  }
  int w = tid >> 5;
  if ((tid & 31) == 0) { red[0][w] = s; red[1][w] = ss; }
  __syncthreads();
  float ts = 0.f, tss = 0.f;
#pragma unroll
  for (int i = 0; i < 8; ++i) { ts += red[0][i]; tss += red[1][i]; }
  float mean = ts * (1.0f / DIM);
  float var  = tss * (1.0f / DIM) - mean * mean;
  float inv  = rsqrtf(var + 1e-5f);
#pragma unroll
  for (int i = 0; i < 4; ++i) {
    int c = tid + 256 * i;
    float o = (vals[i] - mean) * inv * g[c] + beta[c];
    xn[(size_t)row * DIM + c] = (__bf16)o;
  }
}

// ---------------------------------------------------------------------------
// Kernel 2: fp32 -> bf16 weight conversion
// ---------------------------------------------------------------------------
__global__ void cvt_bf16_kernel(const float* __restrict__ a, __bf16* __restrict__ o, int n) {
  for (int i = blockIdx.x * blockDim.x + threadIdx.x; i < n; i += gridDim.x * blockDim.x)
    o[i] = (__bf16)a[i];
}

// ---------------------------------------------------------------------------
// Kernel 3: QKV GEMM  xn[8192x1024] @ wq[1024x3072] + bias.
// Block = 4 waves, 64x64 tile. A (64x32) and B (32x64) tiles TDM
// double-buffered in LDS; inner loop batches all ds_loads then issues
// 4 WMMAs back-to-back. Epilogue scatters Q (prescaled 1/8), K transposed,
// and V into attention-friendly layouts.
// ---------------------------------------------------------------------------
__global__ __launch_bounds__(128)
void qkv_gemm_kernel(const __bf16* __restrict__ xn, const __bf16* __restrict__ wq,
                     const float* __restrict__ bias,
                     __bf16* __restrict__ Q, __bf16* __restrict__ Kt,
                     __bf16* __restrict__ V) {
  __shared__ __bf16 ash[2][64 * 32];   // A: 64 rows x 32 k
  __shared__ __bf16 bsh[2][32 * 64];   // B: 32 k   x 64 n
  int lane = threadIdx.x & 31;
  int w    = threadIdx.x >> 5;
  int m0b  = blockIdx.y * 64;
  int n0   = blockIdx.x * 64;
  int koff = (lane >> 4) * 8;
  int arow = w * 16 + (lane & 15);     // A row within block tile

  if (w == 0) {
    tdm_load_2d(lds_off_of(&ash[0][0]), xn + (size_t)m0b * DIM,
                32, 64, DIM, ROWS_ALL, DIM);
    tdm_load_2d(lds_off_of(&bsh[0][0]), wq + n0,
                64, 32, 3 * DIM, DIM, 3 * DIM);
  }
  __builtin_amdgcn_s_wait_tensorcnt(0);   // no-op for waves with TENSORcnt==0
  __syncthreads();

  v8f acc[4] = {};
  int cur = 0;
  for (int k0 = 0; k0 < DIM; k0 += 32) {
    int nxt = cur ^ 1;
    if (w == 0 && (k0 + 32) < DIM) {
      tdm_load_2d(lds_off_of(&ash[nxt][0]), xn + (size_t)m0b * DIM + (k0 + 32),
                  32, 64, DIM, ROWS_ALL, DIM);
      tdm_load_2d(lds_off_of(&bsh[nxt][0]), wq + (size_t)(k0 + 32) * (3 * DIM) + n0,
                  64, 32, 3 * DIM, DIM, 3 * DIM);
    }
    ABu a, bt[4];
    a.u[0] = *(const u32x4*)(&ash[cur][arow * 32 + koff]);
    a.u[1] = *(const u32x4*)(&ash[cur][arow * 32 + koff + 16]);
#pragma unroll
    for (int j = 0; j < 4; ++j) {
#pragma unroll
      for (int s = 0; s < 16; ++s)
        bt[j].v[s] = bsh[cur][(koff + s + (s >= 8 ? 8 : 0)) * 64 + 16 * j + (lane & 15)];
    }
#pragma unroll
    for (int j = 0; j < 4; ++j) acc[j] = wmma_bf16(a.v, bt[j].v, acc[j]);
    __builtin_amdgcn_s_wait_tensorcnt(0);
    __syncthreads();
    cur = nxt;
  }

  int m0 = m0b + w * 16;
  int mh = (lane >> 4) * 8;
  int nl = lane & 15;
#pragma unroll
  for (int j = 0; j < 4; ++j) {
#pragma unroll
    for (int r = 0; r < 8; ++r) {
      int m = m0 + mh + r;
      int n = n0 + 16 * j + nl;
      float val = acc[j][r] + bias[n];
      int b = (m >> 11) & 1;          // batch
      int t = m & 2047;               // position within batch
      bool isCtx = (m >= ROWS_X);
      int seg = n >> 10;              // 0=Q 1=K 2=V
      int nn  = n & 1023;
      int h = nn >> 6, d = nn & 63;
      int bh = b * HEADS + h;
      if (seg == 0) {
        if (!isCtx) Q[((size_t)bh * TT + t) * HDIM + d] = (__bf16)(val * 0.125f);
      } else {
        int srow = isCtx ? (TT + t) : t;
        if (seg == 1) Kt[((size_t)bh * HDIM + d) * LTOT + srow] = (__bf16)val;
        else          V [((size_t)bh * LTOT + srow) * HDIM + d] = (__bf16)val;
      }
    }
  }
}

// ---------------------------------------------------------------------------
// Kernel 4: flash attention. Block = 4 waves sharing one (b,h); each wave owns
// a 16-row Q tile. K (64d x 32s) and V (32s x 64d) tiles TDM double-buffered
// in LDS, shared by all 4 waves. Online softmax with shfl reductions;
// P reshuffled C->A layout through per-wave LDS.
// ---------------------------------------------------------------------------
__global__ __launch_bounds__(128)
void attn_kernel(const __bf16* __restrict__ Q, const __bf16* __restrict__ Kt,
                 const __bf16* __restrict__ V, __bf16* __restrict__ ao) {
  __shared__ __bf16 ksh[2][64 * 32];   // rows d, 32 s cols
  __shared__ __bf16 vsh[2][32 * 64];   // rows s, 64 d cols
  __shared__ __bf16 psh[4][16 * 32];   // per-wave P reshuffle scratch
  int lane = threadIdx.x & 31;
  int w    = threadIdx.x >> 5;
  int b = blockIdx.z, h = blockIdx.y;
  int bh = b * HEADS + h;
  int t0 = blockIdx.x * 64 + w * 16;
  int koff = (lane >> 4) * 8;
  int nl   = lane & 15;
  int mh   = (lane >> 4) * 8;

  // Q tile: 16 rows x 64 d, two 16x32 A operands (d in [0,32) and [32,64))
  ABu qa[2];
  const __bf16* qrow = Q + ((size_t)bh * TT + t0 + nl) * HDIM;
#pragma unroll
  for (int kk = 0; kk < 2; ++kk) {
    qa[kk].u[0] = *(const u32x4*)(qrow + 32 * kk + koff);
    qa[kk].u[1] = *(const u32x4*)(qrow + 32 * kk + koff + 16);
  }

  const __bf16* kbase = Kt + (size_t)bh * HDIM * LTOT;
  const __bf16* vbase = V  + (size_t)bh * LTOT * HDIM;
  if (w == 0) {
    tdm_load_2d(lds_off_of(&ksh[0][0]), kbase, 32, 64, LTOT, HDIM, LTOT);
    tdm_load_2d(lds_off_of(&vsh[0][0]), vbase, 64, 32, HDIM, LTOT, HDIM);
  }
  __builtin_amdgcn_s_wait_tensorcnt(0);
  __syncthreads();

  float mrow[8], lrow[8];
#pragma unroll
  for (int r = 0; r < 8; ++r) { mrow[r] = -3.0e38f; lrow[r] = 0.f; }
  v8f o[4] = {};

  __bf16* ps = &psh[w][0];
  int cur = 0;

  for (int s0 = 0; s0 < LTOT; s0 += 32) {
    int nxt = cur ^ 1;
    if (w == 0 && (s0 + 32) < LTOT) {
      tdm_load_2d(lds_off_of(&ksh[nxt][0]), kbase + (s0 + 32), 32, 64, LTOT, HDIM, LTOT);
      tdm_load_2d(lds_off_of(&vsh[nxt][0]), vbase + (size_t)(s0 + 32) * HDIM, 64, 32, HDIM, LTOT, HDIM);
    }
    // batch all four K B-operand loads, then 4 score WMMAs back-to-back
    ABu kb[4];
#pragma unroll
    for (int kk = 0; kk < 2; ++kk) {
#pragma unroll
      for (int s = 0; s < 16; ++s) {
        const __bf16* kp = &ksh[cur][(32 * kk + koff + s + (s >= 8 ? 8 : 0)) * 32 + nl];
        kb[2 * kk + 0].v[s] = kp[0];
        kb[2 * kk + 1].v[s] = kp[16];
      }
    }
    v8f sc0 = {}, sc1 = {};
    sc0 = wmma_bf16(qa[0].v, kb[0].v, sc0);
    sc1 = wmma_bf16(qa[0].v, kb[1].v, sc1);
    sc0 = wmma_bf16(qa[1].v, kb[2].v, sc0);
    sc1 = wmma_bf16(qa[1].v, kb[3].v, sc1);
    // online softmax per output row (row r+8*(lane>=16) lives in a 16-lane half)
#pragma unroll
    for (int r = 0; r < 8; ++r) {
      float e0 = sc0[r], e1 = sc1[r];
      float mx = fmaxf(e0, e1);
      mx = fmaxf(mx, __shfl_xor(mx, 1, 32));
      mx = fmaxf(mx, __shfl_xor(mx, 2, 32));
      mx = fmaxf(mx, __shfl_xor(mx, 4, 32));
      mx = fmaxf(mx, __shfl_xor(mx, 8, 32));
      float mn  = fmaxf(mrow[r], mx);
      float fac = __expf(mrow[r] - mn);
      float p0  = __expf(e0 - mn);
      float p1  = __expf(e1 - mn);
      float psum = p0 + p1;
      psum += __shfl_xor(psum, 1, 32);
      psum += __shfl_xor(psum, 2, 32);
      psum += __shfl_xor(psum, 4, 32);
      psum += __shfl_xor(psum, 8, 32);
      lrow[r] = lrow[r] * fac + psum;
      mrow[r] = mn;
      o[0][r] *= fac; o[1][r] *= fac; o[2][r] *= fac; o[3][r] *= fac;
      int m = mh + r;                       // C-layout row
      ps[m * 32 + nl]      = (__bf16)p0;    // cols [0,16)
      ps[m * 32 + nl + 16] = (__bf16)p1;    // cols [16,32)
    }
    asm volatile("s_wait_dscnt 0x0" ::: "memory");
    ABu pa;                                 // P in A-layout (16x32 over local s)
    const __bf16* pr = ps + nl * 32 + koff;
    pa.u[0] = *(const u32x4*)(pr);
    pa.u[1] = *(const u32x4*)(pr + 16);
    // batch all four V B-operand loads, then 4 PV WMMAs back-to-back
    ABu vb[4];
#pragma unroll
    for (int j = 0; j < 4; ++j) {
#pragma unroll
      for (int s = 0; s < 16; ++s)
        vb[j].v[s] = vsh[cur][(koff + s + (s >= 8 ? 8 : 0)) * 64 + 16 * j + nl];
    }
#pragma unroll
    for (int j = 0; j < 4; ++j) o[j] = wmma_bf16(pa.v, vb[j].v, o[j]);
    __builtin_amdgcn_s_wait_tensorcnt(0);
    __syncthreads();
    cur = nxt;
  }

#pragma unroll
  for (int r = 0; r < 8; ++r) {
    float inv = 1.0f / lrow[r];
    int t = t0 + mh + r;
#pragma unroll
    for (int j = 0; j < 4; ++j) {
      int col = h * HDIM + 16 * j + nl;
      ao[((size_t)b * TT + t) * DIM + col] = (__bf16)(o[j][r] * inv);
    }
  }
}

// ---------------------------------------------------------------------------
// Kernel 5: output projection  ao[4096x1024] @ wo[1024x1024] + b_out + x,
// TDM double-buffered like kernel 3.
// ---------------------------------------------------------------------------
__global__ __launch_bounds__(128)
void oproj_kernel(const __bf16* __restrict__ ao, const __bf16* __restrict__ wo,
                  const float* __restrict__ bias, const float* __restrict__ x,
                  float* __restrict__ out) {
  __shared__ __bf16 ash[2][64 * 32];
  __shared__ __bf16 bsh[2][32 * 64];
  int lane = threadIdx.x & 31;
  int w    = threadIdx.x >> 5;
  int m0b  = blockIdx.y * 64;
  int n0   = blockIdx.x * 64;
  int koff = (lane >> 4) * 8;
  int arow = w * 16 + (lane & 15);

  if (w == 0) {
    tdm_load_2d(lds_off_of(&ash[0][0]), ao + (size_t)m0b * DIM,
                32, 64, DIM, ROWS_X, DIM);
    tdm_load_2d(lds_off_of(&bsh[0][0]), wo + n0,
                64, 32, DIM, DIM, DIM);
  }
  __builtin_amdgcn_s_wait_tensorcnt(0);
  __syncthreads();

  v8f acc[4] = {};
  int cur = 0;
  for (int k0 = 0; k0 < DIM; k0 += 32) {
    int nxt = cur ^ 1;
    if (w == 0 && (k0 + 32) < DIM) {
      tdm_load_2d(lds_off_of(&ash[nxt][0]), ao + (size_t)m0b * DIM + (k0 + 32),
                  32, 64, DIM, ROWS_X, DIM);
      tdm_load_2d(lds_off_of(&bsh[nxt][0]), wo + (size_t)(k0 + 32) * DIM + n0,
                  64, 32, DIM, DIM, DIM);
    }
    ABu a, bt[4];
    a.u[0] = *(const u32x4*)(&ash[cur][arow * 32 + koff]);
    a.u[1] = *(const u32x4*)(&ash[cur][arow * 32 + koff + 16]);
#pragma unroll
    for (int j = 0; j < 4; ++j) {
#pragma unroll
      for (int s = 0; s < 16; ++s)
        bt[j].v[s] = bsh[cur][(koff + s + (s >= 8 ? 8 : 0)) * 64 + 16 * j + (lane & 15)];
    }
#pragma unroll
    for (int j = 0; j < 4; ++j) acc[j] = wmma_bf16(a.v, bt[j].v, acc[j]);
    __builtin_amdgcn_s_wait_tensorcnt(0);
    __syncthreads();
    cur = nxt;
  }

  int m0 = m0b + w * 16;
  int mh = (lane >> 4) * 8, nl = lane & 15;
#pragma unroll
  for (int j = 0; j < 4; ++j) {
#pragma unroll
    for (int r = 0; r < 8; ++r) {
      int m = m0 + mh + r;
      int n = n0 + 16 * j + nl;
      out[(size_t)m * DIM + n] = acc[j][r] + bias[n] + x[(size_t)m * DIM + n];
    }
  }
}

// ---------------------------------------------------------------------------
extern "C" void kernel_launch(void* const* d_in, const int* in_sizes, int n_in,
                              void* d_out, int out_size, void* d_ws, size_t ws_size,
                              hipStream_t stream) {
  const float* x     = (const float*)d_in[0];
  const float* ctx   = (const float*)d_in[1];
  const float* w_qkv = (const float*)d_in[2];
  const float* b_qkv = (const float*)d_in[3];
  const float* w_out = (const float*)d_in[4];
  const float* b_out = (const float*)d_in[5];
  const float* ln_g  = (const float*)d_in[6];
  const float* ln_b  = (const float*)d_in[7];
  float* out = (float*)d_out;

  char* ws = (char*)d_ws;
  size_t off = 0;
  __bf16* xn = (__bf16*)(ws + off); off += (size_t)ROWS_ALL * DIM * 2;          // 16 MB
  __bf16* wq = (__bf16*)(ws + off); off += (size_t)DIM * 3 * DIM * 2;           //  6 MB
  __bf16* wo = (__bf16*)(ws + off); off += (size_t)DIM * DIM * 2;               //  2 MB
  __bf16* Qs = (__bf16*)(ws + off); off += (size_t)BB * HEADS * TT * HDIM * 2;  //  8 MB
  __bf16* Kt = (__bf16*)(ws + off); off += (size_t)BB * HEADS * HDIM * LTOT * 2;// 16 MB
  __bf16* Vs = (__bf16*)(ws + off); off += (size_t)BB * HEADS * LTOT * HDIM * 2;// 16 MB
  __bf16* ao = xn; // alias: xn fully consumed by qkv_gemm before attn writes ao

  ln_bf16_kernel<<<ROWS_ALL, 256, 0, stream>>>(x, ctx, ln_g, ln_b, xn);
  cvt_bf16_kernel<<<1024, 256, 0, stream>>>(w_qkv, wq, DIM * 3 * DIM);
  cvt_bf16_kernel<<<512, 256, 0, stream>>>(w_out, wo, DIM * DIM);
  qkv_gemm_kernel<<<dim3(48, 128), 128, 0, stream>>>(xn, wq, b_qkv, Qs, Kt, Vs);
  attn_kernel<<<dim3(TT / 64, HEADS, BB), 128, 0, stream>>>(Qs, Kt, Vs, ao);
  oproj_kernel<<<dim3(DIM / 64, ROWS_X / 64), 128, 0, stream>>>(ao, wo, b_out, x, out);
}
